// CrossAttention_30777735643440
// MI455X (gfx1250) — matrix operand order repaired and checked by reading, not compiled, not run on hardware
//
#include <hip/hip_runtime.h>
#include <hip/hip_bf16.h>
#include <math.h>

typedef __attribute__((ext_vector_type(16))) _Float16 v16h;
typedef __attribute__((ext_vector_type(8)))  _Float16 v8h;
typedef __attribute__((ext_vector_type(8)))  float    v8f;

// Problem constants (B=1, DIM=64, RED=2, H=W=192, num_cats=2)
#define DIMC   64
#define CM     32          // match channels = DIM/RED
#define HF     192
#define WF     192
#define HWF    (HF*WF)     // 36864
#define HH     96          // pooled H
#define WW1    96          // pooled W (x side)
#define WW2K   480         // 96 * (1 + 2*num_cats)
#define KDIM   288         // 9 * 32 unfold-channel dim
#define WADJ6  960         // 192 * 5

// ---------------- conv 1x1 over 64 input channels ----------------
__global__ void k_conv1x1(const float* __restrict__ in, const float* __restrict__ w,
                          const float* __restrict__ b, float* __restrict__ out, int Cout) {
    int t = blockIdx.x * blockDim.x + threadIdx.x;
    int total = Cout * HWF;
    if (t >= total) return;
    int co = t / HWF;
    int p  = t - co * HWF;
    float acc = b[co];
    const float* wr = w + co * DIMC;
    #pragma unroll 8
    for (int ci = 0; ci < DIMC; ci++)
        acc += wr[ci] * in[ci * HWF + p];
    out[t] = acc;
}

// ---------------- depthwise 3x3, zero-pad SAME ----------------
__global__ void k_dw3(const float* __restrict__ in, const float* __restrict__ w,
                      const float* __restrict__ b, float* __restrict__ out, int C) {
    int t = blockIdx.x * blockDim.x + threadIdx.x;
    int total = C * HWF;
    if (t >= total) return;
    int c = t / HWF;
    int p = t - c * HWF;
    int h = p / WF, x = p - h * WF;
    float acc = b[c];
    const float* wk = w + c * 9;
    #pragma unroll
    for (int ki = 0; ki < 3; ki++) {
        int hh = h + ki - 1;
        if (hh < 0 || hh >= HF) continue;
        #pragma unroll
        for (int kj = 0; kj < 3; kj++) {
            int ww = x + kj - 1;
            if (ww < 0 || ww >= WF) continue;
            acc += wk[ki * 3 + kj] * in[c * HWF + hh * WF + ww];
        }
    }
    out[t] = acc;
}

// ---------------- exact 2x2 average pool ----------------
__global__ void k_avgpool2(const float* __restrict__ in, float* __restrict__ out, int C) {
    int t = blockIdx.x * blockDim.x + threadIdx.x;
    int total = C * HH * WW1;
    if (t >= total) return;
    int c = t / (HH * WW1);
    int p = t - c * (HH * WW1);
    int h = p / WW1, x = p - h * WW1;
    const float* s = in + c * HWF + (h * 2) * WF + x * 2;
    out[t] = 0.25f * (s[0] + s[1] + s[WF] + s[WF + 1]);
}

// ---------------- adjacent(ym, num_cats=2, stride=1) : [32,96,96] -> [32,96,480] ----------------
__global__ void k_adjacent(const float* __restrict__ ym, float* __restrict__ out) {
    int t = blockIdx.x * blockDim.x + threadIdx.x;
    int total = CM * HH * WW2K;
    if (t >= total) return;
    int c = t / (HH * WW2K);
    int p = t - c * (HH * WW2K);
    int h = p / WW2K, ww = p - h * WW2K;
    int blk = ww / WW1, jj = ww - blk * WW1;
    const int sh[5] = {0, 1, -1, 2, -2};   // orig, roll-1(up), roll+1(down), roll-2, roll+2
    int hs = h + sh[blk];
    hs = (hs % HH + HH) % HH;
    out[t] = ym[c * (HH * WW1) + hs * WW1 + jj];
}

__device__ __forceinline__ int refl(int t, int n) {
    if (t < 0) return -t;
    if (t >= n) return 2 * n - 2 - t;
    return t;
}

// ---------------- reflect-pad 3x3 unfold + L2-normalize -> f16 [h][i][k=288] ----------------
__global__ void k_unfold_l2n(const float* __restrict__ src, int Ws, _Float16* __restrict__ out) {
    int t = blockIdx.x * blockDim.x + threadIdx.x;
    int total = HH * Ws;
    if (t >= total) return;
    int h = t / Ws, i = t - h * Ws;
    // pass 1: sum of squares over the 288-dim
    float ss = 0.f;
    for (int c = 0; c < CM; c++) {
        const float* sc = src + c * HH * Ws;
        #pragma unroll
        for (int ki = 0; ki < 3; ki++) {
            int hh = refl(h - 1 + ki, HH);
            #pragma unroll
            for (int kj = 0; kj < 3; kj++) {
                int ww = refl(i - 1 + kj, Ws);
                float v = sc[hh * Ws + ww];
                ss += v * v;
            }
        }
    }
    float inv = 1.f / fmaxf(sqrtf(ss), 1e-12f);
    // pass 2: normalized f16 write, m = c*9 + ki*3 + kj
    _Float16* o = out + (size_t)t * KDIM;
    for (int c = 0; c < CM; c++) {
        const float* sc = src + c * HH * Ws;
        #pragma unroll
        for (int ki = 0; ki < 3; ki++) {
            int hh = refl(h - 1 + ki, HH);
            #pragma unroll
            for (int kj = 0; kj < 3; kj++) {
                int ww = refl(i - 1 + kj, Ws);
                o[c * 9 + ki * 3 + kj] = (_Float16)(sc[hh * Ws + ww] * inv);
            }
        }
    }
}

// Load one K-group (3 k-steps of 32) of B fragments for this lane.
// B layout (32x16): lanes 0-15 hold K {0..7, 16..23}, lanes 16-31 hold K {8..15, 24..31}, N = lane&15.
__device__ __forceinline__ void load_bgroup(const _Float16* __restrict__ Bm,
                                            int nt, int g, int lr, int kbaseB,
                                            v16h* __restrict__ dst) {
    #pragma unroll
    for (int u = 0; u < 3; u++) {
        int ks = g * 3 + u;
        const _Float16* pb = Bm + (size_t)(nt * 16 + lr) * KDIM + ks * 32 + kbaseB;
        v8h lo = *(const v8h*)(pb);
        v8h hi = *(const v8h*)(pb + 16);
        v16h bv;
        #pragma unroll
        for (int q = 0; q < 8; q++) { bv[q] = lo[q]; bv[8 + q] = hi[q]; }
        dst[u] = bv;
    }
}

// ---------------- sim GEMM per h-row via v_wmma_f32_16x16x32_f16 + fused max/argmax ----------------
// A[i][k] = xu[h][i][k] (96x288)   B[j][k] = yu[h][j][k] (480x288, col-major for B)
// block = one h (96 blocks), 6 waves, wave w owns M-tile w; loops 30 N-tiles x 9 K-steps.
// B is double-buffered in registers at 3-k-step granularity to overlap loads with WMMA.
__global__ void k_sim_wmma(const _Float16* __restrict__ xu, const _Float16* __restrict__ yu,
                           float* __restrict__ score, int* __restrict__ idx) {
    const int h    = blockIdx.x;
    const int wave = threadIdx.x >> 5;
    const int lane = threadIdx.x & 31;
    const int lr   = lane & 15;
    const int half = lane >> 4;

    const _Float16* A  = xu + (size_t)h * WW1 * KDIM;
    const _Float16* Bm = yu + (size_t)h * WW2K * KDIM;

    // --- load all 9 A k-step fragments (ISA 16-bit A 16x32 layout) ---
    // lanes 0-15: K in {0..7, 16..23}; lanes 16-31: K in {8..15, 24..31}; row M = lane&15
    const int arow   = wave * 16 + lr;
    const int kbaseA = half * 8;
    v16h a[9];
    #pragma unroll
    for (int ks = 0; ks < 9; ks++) {
        const _Float16* pa = A + (size_t)arow * KDIM + ks * 32 + kbaseA;
        v8h lo = *(const v8h*)(pa);
        v8h hi = *(const v8h*)(pa + 16);
        v16h av;
        #pragma unroll
        for (int q = 0; q < 8; q++) { av[q] = lo[q]; av[8 + q] = hi[q]; }
        a[ks] = av;
    }

    float bestV[8];
    int   bestJ[8];
    #pragma unroll
    for (int r = 0; r < 8; r++) { bestV[r] = -3.4e38f; bestJ[r] = 0; }

    const int kbaseB = half * 8;

    v16h bb[2][3];
    load_bgroup(Bm, 0, 0, lr, kbaseB, bb[0]);   // prime the pipeline

    v8f acc = {0.f, 0.f, 0.f, 0.f, 0.f, 0.f, 0.f, 0.f};
    int nt = 0, g = 0, buf = 0;
    // 90 groups total (30 N-tiles x 3 groups of 3 k-steps); unroll 6 = lcm(buf parity, group cycle)
    #pragma unroll 6
    for (int gi = 0; gi < 90; gi++) {
        int nnt = nt, ng = g + 1;
        if (ng == 3) { ng = 0; nnt = nt + 1; }
        if (gi + 1 < 90)
            load_bgroup(Bm, nnt, ng, lr, kbaseB, bb[buf ^ 1]);   // prefetch next group
        #pragma unroll
        for (int u = 0; u < 3; u++)
            acc = __builtin_amdgcn_wmma_f32_16x16x32_f16(false, a[g * 3 + u], false,
                                                         bb[buf][u], (short)0, acc,
                                                         false, false);
        if (g == 2) {
            // C layout: VGPR r -> row M = r + half*8, N = lane&15. Reduce over the 16-lane half.
            #pragma unroll
            for (int r = 0; r < 8; r++) {
                float v = acc[r];
                int   j = nt * 16 + lr;
                #pragma unroll
                for (int d = 8; d > 0; d >>= 1) {   // xor masks 1,2,4,8 stay inside the half
                    float ov = __shfl_xor(v, d, 32);
                    int   oj = __shfl_xor(j, d, 32);
                    int take = (int)(ov > v) | ((int)(ov == v) & (int)(oj < j));
                    v = take ? ov : v;
                    j = take ? oj : j;
                }
                int take = (int)(v > bestV[r]) | ((int)(v == bestV[r]) & (int)(j < bestJ[r]));
                bestV[r] = take ? v : bestV[r];
                bestJ[r] = take ? j : bestJ[r];
            }
            acc = (v8f){0.f, 0.f, 0.f, 0.f, 0.f, 0.f, 0.f, 0.f};
        }
        nt = nnt; g = ng; buf ^= 1;
    }

    if (lr == 0) {
        #pragma unroll
        for (int r = 0; r < 8; r++) {
            int row = wave * 16 + half * 8 + r;
            score[h * WW1 + row] = bestV[r];
            idx[h * WW1 + row]   = bestJ[r];
        }
    }
}

// ---------------- fused gather + fold(6,2,pad=2) + divisor ----------------
// out[c,hh,ww] = sum over valid (ki,kj) of yvu[c*36+ki*6+kj, oh, idx[oh,ow]] * score[oh,ow] / count
// yvu[.., oh, j] = adjacent(yv,2,2) reflect-padded at (2*oh+ki-2, 2*j+kj-2)
__global__ void k_fold_out(const float* __restrict__ yv, const float* __restrict__ score,
                           const int* __restrict__ idx, float* __restrict__ out) {
    int t = blockIdx.x * blockDim.x + threadIdx.x;
    int total = DIMC * HWF;
    if (t >= total) return;
    int c  = t / HWF;
    int p  = t - c * HWF;
    int hh = p / WF, ww = p - hh * WF;

    const int sh6[5] = {0, 2, -2, 4, -4};
    float acc = 0.f;
    int cnt = 0;
    int ki0 = hh & 1;   // ki must match parity of hh (ki = hh+2-2*oh)
    int kj0 = ww & 1;
    for (int ki = ki0; ki < 6; ki += 2) {
        int oh2 = hh + 2 - ki;
        if (oh2 < 0) continue;
        int oh = oh2 >> 1;
        if (oh >= HH) continue;
        for (int kj = kj0; kj < 6; kj += 2) {
            int ow2 = ww + 2 - kj;
            if (ow2 < 0) continue;
            int ow = ow2 >> 1;
            if (ow >= WW1) continue;
            cnt++;
            int q = oh * WW1 + ow;
            int j = idx[q];
            float s = score[q];
            int hp = oh * 2 + ki - 2;                 // reflect into [0,192)
            if (hp < 0) hp = -hp; else if (hp >= HF) hp = 2 * HF - 2 - hp;
            int wp = j * 2 + kj - 2;                  // reflect into [0,960)
            if (wp < 0) wp = -wp; else if (wp >= WADJ6) wp = 2 * WADJ6 - 2 - wp;
            int blk = wp / WF, jj = wp - blk * WF;
            int hv = hp + sh6[blk];
            if (hv < 0) hv += HF; else if (hv >= HF) hv -= HF;
            acc += yv[c * HWF + hv * WF + jj] * s;
        }
    }
    out[t] = acc / (float)cnt;
}

// ---------------- workspace layout (bytes) ----------------
#define OFF_T1    0                                   // 9.44 MB  (also reused for ym_adj)
#define OFF_T2    9437184                             // 9.44 MB  (also reused for xu f16)
#define OFF_XM    18874368                            // 1.18 MB
#define OFF_YM    20054016                            // 1.18 MB
#define OFF_YV    21233664                            // 9.44 MB
#define OFF_YU    30670848                            // 26.5 MB (f16)
#define OFF_SC    57212928                            // 36 KB
#define OFF_IDX   57249792                            // 36 KB

extern "C" void kernel_launch(void* const* d_in, const int* in_sizes, int n_in,
                              void* d_out, int out_size, void* d_ws, size_t ws_size,
                              hipStream_t stream) {
    const float* x   = (const float*)d_in[0];
    const float* y   = (const float*)d_in[1];
    const float* mw1 = (const float*)d_in[2];
    const float* mb1 = (const float*)d_in[3];
    const float* mw2 = (const float*)d_in[4];
    const float* mb2 = (const float*)d_in[5];
    const float* vw1 = (const float*)d_in[6];
    const float* vb1 = (const float*)d_in[7];
    const float* vw2 = (const float*)d_in[8];
    const float* vb2 = (const float*)d_in[9];
    (void)in_sizes; (void)n_in; (void)ws_size; (void)out_size;

    char* ws = (char*)d_ws;
    float*     t1    = (float*)(ws + OFF_T1);
    float*     t2    = (float*)(ws + OFF_T2);
    float*     xm    = (float*)(ws + OFF_XM);
    float*     ym    = (float*)(ws + OFF_YM);
    float*     yv    = (float*)(ws + OFF_YV);
    float*     yadj  = (float*)(ws + OFF_T1);   // reuse t1 after value path done
    _Float16*  xu    = (_Float16*)(ws + OFF_T2); // reuse t2 after match path done
    _Float16*  yu    = (_Float16*)(ws + OFF_YU);
    float*     score = (float*)(ws + OFF_SC);
    int*       idxb  = (int*)(ws + OFF_IDX);

    const int TB = 256;
    auto nb = [](int n, int b) { return (n + b - 1) / b; };

    // match path: x -> xm
    k_conv1x1 <<<nb(CM * HWF, TB), TB, 0, stream>>>(x, mw1, mb1, t1, CM);
    k_dw3     <<<nb(CM * HWF, TB), TB, 0, stream>>>(t1, mw2, mb2, t2, CM);
    k_avgpool2<<<nb(CM * HH * WW1, TB), TB, 0, stream>>>(t2, xm, CM);
    // match path: y -> ym
    k_conv1x1 <<<nb(CM * HWF, TB), TB, 0, stream>>>(y, mw1, mb1, t1, CM);
    k_dw3     <<<nb(CM * HWF, TB), TB, 0, stream>>>(t1, mw2, mb2, t2, CM);
    k_avgpool2<<<nb(CM * HH * WW1, TB), TB, 0, stream>>>(t2, ym, CM);
    // value path: y -> yv
    k_conv1x1 <<<nb(DIMC * HWF, TB), TB, 0, stream>>>(y, vw1, vb1, t1, DIMC);
    k_dw3     <<<nb(DIMC * HWF, TB), TB, 0, stream>>>(t1, vw2, vb2, yv, DIMC);
    // adjacency + unfold/normalize (f16 for WMMA)
    k_adjacent  <<<nb(CM * HH * WW2K, TB), TB, 0, stream>>>(ym, yadj);
    k_unfold_l2n<<<nb(HH * WW1, TB), TB, 0, stream>>>(xm, WW1, xu);
    k_unfold_l2n<<<nb(HH * WW2K, TB), TB, 0, stream>>>(yadj, WW2K, yu);
    // similarity GEMM (WMMA) + argmax
    k_sim_wmma<<<HH, 192, 0, stream>>>(xu, yu, score, idxb);
    // fused gather + fold + divisor
    k_fold_out<<<nb(DIMC * HWF, TB), TB, 0, stream>>>(yv, score, idxb, (float*)d_out);
}
